// MGCN_TPN_Classifier_union_gai_55310588838068
// MI455X (gfx1250) — compile-verified, matrix operand order checked by reading, not run on hardware
//
#include <hip/hip_runtime.h>
#include <hip/hip_bf16.h>
#include <stdint.h>

// ---------------------------------------------------------------------------
// Problem constants (from reference)
// ---------------------------------------------------------------------------
#define NUM_SHOT    1000
#define NUM_QUERY   6400
#define NUM_TOTAL   (NUM_SHOT + NUM_QUERY)   // 7400
#define DIM         256
#define NUM_IMGS    32
#define IMG_SIZE    512
#define SP_PER_IMG  200
#define CLS         5
#define CLS_PAD     16                       // padded class dim for WMMA N

#define OUT0_ELEMS  ((size_t)NUM_IMGS * IMG_SIZE * IMG_SIZE * CLS)  // 41,943,040
#define PIX_PER_IMG (IMG_SIZE * IMG_SIZE)                            // 262144
#define FLT_PER_IMG (PIX_PER_IMG * CLS)                              // 1,310,720

// workspace layout (float offsets)
#define WS_SUMS    0                 // 5*256
#define WS_COUNTS  1280              // 5
#define WS_PNORM   1296              // 16
#define WS_PROTO   1312              // 16*256  (16B aligned: 1312*4 = 5248)
#define WS_QNORM   (1312 + 16*256)   // 6400
#define WS_ZERO_BYTES (1312 * 4)     // zero sums+counts+pnorm each call

typedef __attribute__((ext_vector_type(16))) _Float16 v16h;
typedef __attribute__((ext_vector_type(8)))  float    v8f;

// ---------------------------------------------------------------------------
// Kernel 1: weighted per-class feature sums + counts (LDS-staged atomics)
// ---------------------------------------------------------------------------
__global__ void accum_kernel(const float* __restrict__ feat,
                             const int* __restrict__ shot_lbl,
                             const int* __restrict__ pred_lbl,
                             const uint8_t* __restrict__ mask,
                             float* __restrict__ sums,    // [5][256]
                             float* __restrict__ counts)  // [5]
{
    __shared__ float lsum[CLS * DIM];
    __shared__ float lcnt[CLS];
    const int t = threadIdx.x;                 // owns dim t
    #pragma unroll
    for (int c = 0; c < CLS; ++c) lsum[c * DIM + t] = 0.0f;
    if (t < CLS) lcnt[t] = 0.0f;
    __syncthreads();

    const int chunk = (NUM_TOTAL + gridDim.x - 1) / gridDim.x;
    const int r0 = blockIdx.x * chunk;
    const int r1 = min(r0 + chunk, NUM_TOTAL);
    for (int r = r0; r < r1; ++r) {
        int lbl;
        float w;
        if (r < NUM_SHOT) { lbl = shot_lbl[r]; w = 1.0f; }
        else              { lbl = pred_lbl[r - NUM_SHOT];
                            w = mask[r - NUM_SHOT] ? 1.0f : 0.0f; }
        // each thread owns a distinct dim -> race-free within the block
        lsum[lbl * DIM + t] += w * feat[(size_t)r * DIM + t];
        if (t == 0) lcnt[lbl] += w;
    }
    __syncthreads();
    #pragma unroll
    for (int c = 0; c < CLS; ++c) atomicAdd(&sums[c * DIM + t], lsum[c * DIM + t]);
    if (t < CLS) atomicAdd(&counts[t], lcnt[t]);
}

// ---------------------------------------------------------------------------
// Kernel 2: finalize prototypes (padded to 16 classes) + prototype norms
// ---------------------------------------------------------------------------
__global__ void proto_kernel(const float* __restrict__ sums,
                             const float* __restrict__ counts,
                             float* __restrict__ proto,   // [16][256]
                             float* __restrict__ pnorm)   // [16] (pre-zeroed)
{
    const int d = threadIdx.x;                 // 256 threads
    #pragma unroll
    for (int c = 0; c < CLS; ++c) {
        const float cnt = counts[c];
        const float p = (cnt > 0.0f) ? (sums[c * DIM + d] / fmaxf(cnt, 1.0f)) : 0.0f;
        proto[c * DIM + d] = p;
        atomicAdd(&pnorm[c], p * p);
    }
    for (int c = CLS; c < CLS_PAD; ++c) proto[c * DIM + d] = 0.0f;
}

// ---------------------------------------------------------------------------
// Kernel 3: per-query squared norms
// ---------------------------------------------------------------------------
__global__ void qnorm_kernel(const float* __restrict__ feat,  // full table
                             float* __restrict__ qnorm)       // [6400]
{
    const int q = blockIdx.x * blockDim.x + threadIdx.x;
    if (q >= NUM_QUERY) return;
    const float4* row = reinterpret_cast<const float4*>(
        feat + (size_t)(NUM_SHOT + q) * DIM);
    float s = 0.0f;
    #pragma unroll 8
    for (int i = 0; i < DIM / 4; ++i) {
        float4 v = row[i];
        s += v.x * v.x + v.y * v.y + v.z * v.z + v.w * v.w;
    }
    qnorm[q] = s;
}

// ---------------------------------------------------------------------------
// Kernel 4: WMMA GEMM  q(6400x256) x protoT(256x16)  ->  logits -> select
//   One wave computes one 16x16 output tile. 400 tiles = 50 blocks x 8 waves.
//   A/B use the 16-bit 16x32 layout (ISA 7.12.2): lane<16 -> K 0..7,16..23;
//   lane>=16 -> K 8..15,24..31, two halves per VGPR.
// ---------------------------------------------------------------------------
__global__ void wmma_logits_kernel(const float* __restrict__ feat,
                                   const float* __restrict__ proto,  // [16][256]
                                   const float* __restrict__ qnorm,  // [6400]
                                   const float* __restrict__ pnorm,  // [16]
                                   const float* __restrict__ predict_p, // [6400][5]
                                   const uint8_t* __restrict__ mask,    // [6400]
                                   float* __restrict__ pout)            // [6400][5]
{
    const int lane = threadIdx.x & 31;
    const int wave = threadIdx.x >> 5;
    const int tile = blockIdx.x * 8 + wave;    // 0..399, exact

    const int m    = lane & 15;
    const int half = lane >> 4;                // 0 or 1
    const int k0   = half * 8;
    const int qrow = tile * 16 + m;            // query row for A loads

    const float* arow = feat + (size_t)(NUM_SHOT + qrow) * DIM;
    const float* brow = proto + (size_t)m * DIM;   // B: n = lane&15

    v8f acc = {};
    #pragma unroll
    for (int kb = 0; kb < DIM; kb += 32) {
        const float4* ap = reinterpret_cast<const float4*>(arow + kb + k0);
        const float4* bp = reinterpret_cast<const float4*>(brow + kb + k0);
        float4 a0 = ap[0], a1 = ap[1], a2 = ap[4], a3 = ap[5];  // K: +0..7, +16..23
        float4 b0 = bp[0], b1 = bp[1], b2 = bp[4], b3 = bp[5];
        v16h a, b;
        a[0]=(_Float16)a0.x; a[1]=(_Float16)a0.y; a[2]=(_Float16)a0.z; a[3]=(_Float16)a0.w;
        a[4]=(_Float16)a1.x; a[5]=(_Float16)a1.y; a[6]=(_Float16)a1.z; a[7]=(_Float16)a1.w;
        a[8]=(_Float16)a2.x; a[9]=(_Float16)a2.y; a[10]=(_Float16)a2.z; a[11]=(_Float16)a2.w;
        a[12]=(_Float16)a3.x; a[13]=(_Float16)a3.y; a[14]=(_Float16)a3.z; a[15]=(_Float16)a3.w;
        b[0]=(_Float16)b0.x; b[1]=(_Float16)b0.y; b[2]=(_Float16)b0.z; b[3]=(_Float16)b0.w;
        b[4]=(_Float16)b1.x; b[5]=(_Float16)b1.y; b[6]=(_Float16)b1.z; b[7]=(_Float16)b1.w;
        b[8]=(_Float16)b2.x; b[9]=(_Float16)b2.y; b[10]=(_Float16)b2.z; b[11]=(_Float16)b2.w;
        b[12]=(_Float16)b3.x; b[13]=(_Float16)b3.y; b[14]=(_Float16)b3.z; b[15]=(_Float16)b3.w;
        acc = __builtin_amdgcn_wmma_f32_16x16x32_f16(
            /*neg_a=*/false, a, /*neg_b=*/false, b,
            /*c_mod=*/(short)0, acc, /*reuse_a=*/false, /*reuse_b=*/false);
    }

    // D layout: VGPR r -> M = r + 8*half ; N = lane&15
    const int n = lane & 15;
    if (n < CLS) {
        const float pn = pnorm[n];
        #pragma unroll
        for (int r = 0; r < 8; ++r) {
            const int mrow = r + 8 * half;
            const int row  = tile * 16 + mrow;           // < 6400 always
            const float dot = acc[r];
            const float logit = -(qnorm[row] + pn - 2.0f * dot);
            const float val = mask[row] ? predict_p[(size_t)row * CLS + n] : logit;
            pout[(size_t)row * CLS + n] = val;
        }
    }
}

// ---------------------------------------------------------------------------
// Kernel 5: gather-expand (the bandwidth-dominant pass)
//   grid (64, 32): blockIdx.y = image, 64 chunks per image.
//   Per-image 200x5 table staged in LDS; one thread per OUTPUT FLOAT
//   -> perfectly coalesced 4B stores (128 B per wave32 store).
// ---------------------------------------------------------------------------
__global__ void expand_kernel(const float* __restrict__ pout,     // [6400][5]
                              const int* __restrict__ segments,   // [32][512][512]
                              float* __restrict__ out)            // [32][512][512][5]
{
    __shared__ float tbl[SP_PER_IMG * CLS];   // 1000 floats
    const int img = blockIdx.y;
    for (int i = threadIdx.x; i < SP_PER_IMG * CLS; i += blockDim.x)
        tbl[i] = pout[(size_t)img * SP_PER_IMG * CLS + i];
    __syncthreads();

    const int* seg = segments + (size_t)img * PIX_PER_IMG;
    float* o = out + (size_t)img * FLT_PER_IMG;

    const unsigned chunk = FLT_PER_IMG / 64;               // 20480
    const unsigned base  = blockIdx.x * chunk;
    #pragma unroll 4
    for (unsigned j = 0; j < chunk; j += blockDim.x) {
        const unsigned f   = base + j + threadIdx.x;       // output float index
        const unsigned pix = f / 5u;                       // mul-hi, constant div
        const unsigned c   = f - pix * 5u;
        const int s = seg[pix];
        o[f] = tbl[s * CLS + c];
    }
}

// ---------------------------------------------------------------------------
// Launch
// ---------------------------------------------------------------------------
extern "C" void kernel_launch(void* const* d_in, const int* in_sizes, int n_in,
                              void* d_out, int out_size, void* d_ws, size_t ws_size,
                              hipStream_t stream) {
    const float*   feat      = (const float*)d_in[0];
    const int*     shot_lbl  = (const int*)d_in[1];
    const float*   predict_p = (const float*)d_in[2];
    const int*     pred_lbl  = (const int*)d_in[3];
    const uint8_t* mask      = (const uint8_t*)d_in[4];   // jnp bool = 1 byte
    const int*     segments  = (const int*)d_in[5];
    // d_in[6] = class_num (constant 5)

    float* ws   = (float*)d_ws;
    float* sums   = ws + WS_SUMS;
    float* counts = ws + WS_COUNTS;
    float* pnorm  = ws + WS_PNORM;
    float* proto  = ws + WS_PROTO;
    float* qnorm  = ws + WS_QNORM;

    float* out  = (float*)d_out;                 // (32,512,512,5)
    float* pout = out + OUT0_ELEMS;              // (6400,5) second tuple output

    hipMemsetAsync(d_ws, 0, WS_ZERO_BYTES, stream);

    accum_kernel<<<64, 256, 0, stream>>>(feat, shot_lbl, pred_lbl, mask, sums, counts);
    proto_kernel<<<1, 256, 0, stream>>>(sums, counts, proto, pnorm);
    qnorm_kernel<<<(NUM_QUERY + 255) / 256, 256, 0, stream>>>(feat, qnorm);
    wmma_logits_kernel<<<50, 256, 0, stream>>>(feat, proto, qnorm, pnorm,
                                               predict_p, mask, pout);
    expand_kernel<<<dim3(64, NUM_IMGS), 256, 0, stream>>>(pout, segments, out);
}
